// InstanceComm_29815662969428
// MI455X (gfx1250) — matrix operand order repaired and checked by reading, not compiled, not run on hardware
//
#include <hip/hip_runtime.h>
#include <hip/hip_bf16.h>

// ---------------- constants ----------------
#define D 128
#define RANGE1 10000000   // vkey space: 8*250*250*20
#define RANGE2 5000000    // akey space: 4*250*250*20
#define VXYZ 1250000
#define SCAN_BLOCK 256
#define SCAN_ITEMS 16
#define SCAN_TILE 4096
#define GA_LDA 72         // 64 + 8 pad (multiple of 8 -> 16B-aligned rows)
#define GBT_LDB 72        // transposed-B row stride, 64 + 8 pad

typedef _Float16 v16h __attribute__((ext_vector_type(16)));
typedef _Float16 h8   __attribute__((ext_vector_type(8)));
typedef _Float16 h4   __attribute__((ext_vector_type(4)));
typedef float    v8f  __attribute__((ext_vector_type(8)));

// ---------------- WMMA fragment loaders (contiguous -> ds_load_b128) ----------------
// A (16-bit 16x32, ISA 7.12.2): lane m=lane&15, half=lane>>4:
//   elements e=0..7  <-> As[m*lda + half*8 + e]        (contiguous 16B)
//   elements e=8..15 <-> As[m*lda + 16 + half*8 + e-8] (contiguous 16B)
__device__ __forceinline__ v16h load_a_frag(const _Float16* __restrict__ p, int lda,
                                            int mn, int hi) {
  const h8 lo = *(const h8*)(p + mn * lda + hi * 8);
  const h8 hh = *(const h8*)(p + mn * lda + 16 + hi * 8);
  return __builtin_shufflevector(lo, hh, 0, 1, 2, 3, 4, 5, 6, 7,
                                 8, 9, 10, 11, 12, 13, 14, 15);
}
// B (32x16): lane n=lane&15, half=lane>>4: element e<8 <-> B[half*8+e][n], e>=8 <-> B[16+half*8+e-8][n].
// With B stored TRANSPOSED in LDS (Bt[n][k]) this is two contiguous 8-halfword runs starting 16 apart.
__device__ __forceinline__ v16h load_b_frag(const _Float16* __restrict__ p, int ldb,
                                            int mn, int hi) {
  const h8 lo = *(const h8*)(p + mn * ldb + hi * 8);
  const h8 hh = *(const h8*)(p + mn * ldb + 16 + hi * 8);
  return __builtin_shufflevector(lo, hh, 0, 1, 2, 3, 4, 5, 6, 7,
                                 8, 9, 10, 11, 12, 13, 14, 15);
}

// ---------------- scalars slots ----------------
// scal[0]=Kminus1 (atomicMax), scal[1]=U, scal[2]=A, scal[3]=amask count
__global__ void k_init_scal(int* scal) {
  if (threadIdx.x == 0) { scal[0] = -1; scal[1] = 0; scal[2] = 0; scal[3] = 0; }
}

// ---------------- voxel keys + presence flags ----------------
__global__ void k_keys_mark(const float* __restrict__ pc, const int* __restrict__ bidx,
                            int* __restrict__ vkey, unsigned* __restrict__ ranks, int n) {
  int p = blockIdx.x * blockDim.x + threadIdx.x;
  if (p >= n) return;
  float x = pc[p * 3 + 0], y = pc[p * 3 + 1], z = pc[p * 3 + 2];
  int xi = (int)floorf((x - (-50.0f)) / 0.4f);
  int yi = (int)floorf((y - (-50.0f)) / 0.4f);
  int zi = (int)floorf((z - (-5.0f)) / 0.4f);
  int key = ((bidx[p] * 250 + xi) * 250 + yi) * 20 + zi;
  vkey[p] = key;
  ranks[key] = 1u;
}

// ---------------- 3-kernel inclusive scan over flag arrays ----------------
__global__ void k_scan_local(unsigned* __restrict__ data, int len, unsigned* __restrict__ bsums) {
  __shared__ unsigned s[SCAN_BLOCK];
  const int t = threadIdx.x;
  const int base = blockIdx.x * SCAN_TILE + t * SCAN_ITEMS;
  unsigned v[SCAN_ITEMS];
  unsigned sum = 0;
#pragma unroll
  for (int i = 0; i < SCAN_ITEMS; ++i) {
    int idx = base + i;
    v[i] = (idx < len) ? data[idx] : 0u;
    sum += v[i];
  }
  s[t] = sum;
  __syncthreads();
  for (int off = 1; off < SCAN_BLOCK; off <<= 1) {
    unsigned q = (t >= off) ? s[t - off] : 0u;
    __syncthreads();
    s[t] += q;
    __syncthreads();
  }
  unsigned run = s[t] - sum;  // exclusive prefix for this thread
#pragma unroll
  for (int i = 0; i < SCAN_ITEMS; ++i) {
    int idx = base + i;
    run += v[i];
    if (idx < len) data[idx] = run;  // inclusive
  }
  if (t == SCAN_BLOCK - 1) bsums[blockIdx.x] = s[t];
}

__global__ void k_scan_bsums(unsigned* __restrict__ bs, int nb) {
  __shared__ unsigned s[SCAN_BLOCK];
  const int t = threadIdx.x;
  const int per = (nb + SCAN_BLOCK - 1) / SCAN_BLOCK;
  const int beg = t * per;
  const int end = min(beg + per, nb);
  unsigned sum = 0;
  for (int i = beg; i < end; ++i) sum += bs[i];
  s[t] = sum;
  __syncthreads();
  for (int off = 1; off < SCAN_BLOCK; off <<= 1) {
    unsigned q = (t >= off) ? s[t - off] : 0u;
    __syncthreads();
    s[t] += q;
    __syncthreads();
  }
  unsigned run = s[t] - sum;  // exclusive
  for (int i = beg; i < end; ++i) { unsigned x = bs[i]; bs[i] = run; run += x; }
}

__global__ void k_scan_add(unsigned* __restrict__ data, int len, const unsigned* __restrict__ bs) {
  unsigned off = bs[blockIdx.x];
  int base = blockIdx.x * SCAN_TILE + threadIdx.x;
#pragma unroll
  for (int i = 0; i < SCAN_ITEMS; ++i) {
    int idx = base + i * SCAN_BLOCK;
    if (idx < len) data[idx] += off;
  }
}

__global__ void k_total(const unsigned* __restrict__ ranks, int len, int* scal, int slot) {
  if (threadIdx.x == 0 && blockIdx.x == 0) scal[slot] = (int)ranks[len - 1];
}

// ---------------- inverse map + K + foreground counts ----------------
__global__ void k_build_inv(const int* __restrict__ vkey, const unsigned* __restrict__ ranks,
                            const int* __restrict__ fg, int* __restrict__ inv,
                            int* __restrict__ uk, float* __restrict__ cnt,
                            int* scal, int n) {
  int p = blockIdx.x * blockDim.x + threadIdx.x;
  if (p >= n) return;
  int r = (int)ranks[vkey[p]] - 1;  // rank among sorted distinct keys
  inv[p] = r;
  uk[r] = vkey[p];
  if (fg[p]) {
    atomicMax(&scal[0], r);
    atomicAdd(&cnt[r], 1.0f);
  }
}

__global__ void k_scatter_feat(const float* __restrict__ pf, const int* __restrict__ inv,
                               const int* __restrict__ fg, float* __restrict__ csum, int total) {
  int i = blockIdx.x * blockDim.x + threadIdx.x;
  if (i >= total) return;
  int p = i >> 7;
  if (fg[p]) atomicAdd(&csum[inv[p] * D + (i & 127)], pf[i]);
}

__global__ void k_cmean(float* __restrict__ csum, const float* __restrict__ cnt,
                        const int* scal, int total) {
  int i = blockIdx.x * blockDim.x + threadIdx.x;
  if (i >= total) return;
  int r = i >> 7;
  int K = scal[0] + 1;
  if (r >= K) return;
  csum[i] = csum[i] / fmaxf(cnt[r], 1.0f);
}

// ---------------- associate voxel keys ----------------
__global__ void k_akey_mark(const int* __restrict__ uk, int* __restrict__ akey,
                            unsigned* __restrict__ ranks2, const int* scal, int n) {
  int r = blockIdx.x * blockDim.x + threadIdx.x;
  int U = scal[1];
  if (r >= U) return;
  int k = uk[r];
  int b = k / VXYZ;
  int rem = k - b * VXYZ;
  int ak = (b >> 1) * VXYZ + rem;
  akey[r] = ak;
  ranks2[ak] = 1u;
}

__global__ void k_build_ainv(const int* __restrict__ akey, const unsigned* __restrict__ ranks2,
                             int* __restrict__ ainv, const int* scal, int n) {
  int r = blockIdx.x * blockDim.x + threadIdx.x;
  int U = scal[1];
  if (r >= U) return;
  ainv[r] = (int)ranks2[akey[r]] - 1;
}

__global__ void k_acnt(const int* __restrict__ ainv, float* __restrict__ acnt,
                       const int* scal, int n) {
  int r = blockIdx.x * blockDim.x + threadIdx.x;
  int K = scal[0] + 1;
  if (r >= K) return;
  atomicAdd(&acnt[ainv[r]], 1.0f);
}

__global__ void k_asum(const float* __restrict__ cfa, const int* __restrict__ ainv,
                       float* __restrict__ asum, const int* scal, int total) {
  int i = blockIdx.x * blockDim.x + threadIdx.x;
  if (i >= total) return;
  int r = i >> 7;
  int K = scal[0] + 1;
  if (r >= K) return;
  atomicAdd(&asum[ainv[r] * D + (i & 127)], cfa[i]);
}

__global__ void k_amean(float* __restrict__ asum, const float* __restrict__ acnt,
                        const int* scal, int total) {
  int i = blockIdx.x * blockDim.x + threadIdx.x;
  if (i >= total) return;
  int a = i >> 7;
  int A = scal[2];
  if (a >= A) return;
  asum[i] = asum[i] / fmaxf(acnt[a], 1.0f);
}

__global__ void k_amask(const int* __restrict__ inv, const int* __restrict__ ainv,
                        const float* __restrict__ cfa, const float* __restrict__ amean,
                        int* __restrict__ amask, int* scal, int n) {
  int p = blockIdx.x * blockDim.x + threadIdx.x;
  if (p >= n) return;
  int c = inv[p];
  int a = ainv[c];
  float s = 0.0f;
  const float* cr = &cfa[c * D];
  const float* ar = &amean[a * D];
#pragma unroll 4
  for (int d = 0; d < D; ++d) s += cr[d] - ar[d];
  int msk = (s > 0.0f) ? 1 : 0;
  amask[p] = msk;
  if (msk) atomicAdd(&scal[3], 1);
}

// ---------------- GEMM1: cfa = leaky(bn(cmean @ w_cp + b)), rows >= K -> 0 ----------------
__global__ __launch_bounds__(256) void k_gemm_cluster(
    const float* __restrict__ cmean, const float* __restrict__ w,
    const float* __restrict__ bias, const float* __restrict__ g,
    const float* __restrict__ be, const float* __restrict__ m,
    const float* __restrict__ v, float* __restrict__ cfa,
    const int* __restrict__ scal, int n) {
  __shared__ __align__(16) _Float16 As[128 * GA_LDA];
  __shared__ __align__(16) _Float16 Bt[128 * GBT_LDB];  // transposed B chunk
  const int K = scal[0] + 1;
  const int rowBase = blockIdx.x * 128;
  const int t = threadIdx.x;
  const int w32 = t >> 5;
  const int lane = t & 31;
  const int mn = lane & 15, hi = lane >> 4;
  v8f acc[8];
#pragma unroll
  for (int i = 0; i < 8; ++i) acc[i] = v8f{0, 0, 0, 0, 0, 0, 0, 0};
  for (int c = 0; c < 2; ++c) {
    if (c == 0) __builtin_prefetch(w + 64 * D, 0, 0);
    // A chunk: 128 rows x 64 k ; float4 global reads -> h4 LDS stores
#pragma unroll
    for (int i = 0; i < 8; ++i) {
      int idx = i * 256 + t;
      int r = idx >> 4;
      int kk = (idx & 15) * 4;
      int gr = rowBase + r;
      float4 x = make_float4(0.f, 0.f, 0.f, 0.f);
      if (gr < n && gr < K) x = *(const float4*)(cmean + gr * D + c * 64 + kk);
      h4 hh = {(_Float16)x.x, (_Float16)x.y, (_Float16)x.z, (_Float16)x.w};
      *(h4*)(As + r * GA_LDA + kk) = hh;
    }
    // B chunk (transposed in LDS): Bt[col][kk]
#pragma unroll
    for (int i = 0; i < 8; ++i) {
      int idx = i * 256 + t;
      int col = idx & 127;
      int kk = (idx >> 7) * 4;
      float a0 = w[(c * 64 + kk + 0) * D + col];
      float a1 = w[(c * 64 + kk + 1) * D + col];
      float a2 = w[(c * 64 + kk + 2) * D + col];
      float a3 = w[(c * 64 + kk + 3) * D + col];
      h4 hh = {(_Float16)a0, (_Float16)a1, (_Float16)a2, (_Float16)a3};
      *(h4*)(Bt + col * GBT_LDB + kk) = hh;
    }
    __syncthreads();
#pragma unroll
    for (int ki = 0; ki < 2; ++ki) {
      v16h a = load_a_frag(&As[(w32 * 16) * GA_LDA + ki * 32], GA_LDA, mn, hi);
#pragma unroll
      for (int nt = 0; nt < 8; ++nt) {
        v16h b = load_b_frag(&Bt[(nt * 16) * GBT_LDB + ki * 32], GBT_LDB, mn, hi);
        acc[nt] = __builtin_amdgcn_wmma_f32_16x16x32_f16(false, a, false, b, (short)0,
                                                         acc[nt], false, false);
      }
    }
    __syncthreads();
  }
#pragma unroll
  for (int nt = 0; nt < 8; ++nt) {
    int col = nt * 16 + mn;
    float bb = bias[col];
    float sc = g[col] * rsqrtf(v[col] + 1e-5f);
    float mm = m[col], bee = be[col];
#pragma unroll
    for (int r = 0; r < 8; ++r) {
      int row = rowBase + w32 * 16 + r + hi * 8;
      if (row < n) {
        float x = acc[nt][r] + bb;
        x = (x - mm) * sc + bee;
        x = (x >= 0.0f) ? x : 0.1f * x;
        cfa[row * D + col] = (row < K) ? x : 0.0f;
      }
    }
  }
}

// ---------------- GEMM2: out = select(leaky(bn(cat @ w_pf + b)), points_feature) ----------------
__global__ __launch_bounds__(256) void k_gemm_point(
    const float* __restrict__ pf, const float* __restrict__ cfa,
    const float* __restrict__ amean, const int* __restrict__ inv,
    const int* __restrict__ ainv, const float* __restrict__ w,
    const float* __restrict__ bias, const float* __restrict__ g,
    const float* __restrict__ be, const float* __restrict__ m,
    const float* __restrict__ v, const int* __restrict__ amask,
    const int* __restrict__ scal, float* __restrict__ out, int n) {
  __shared__ __align__(16) _Float16 As[128 * GA_LDA];
  __shared__ __align__(16) _Float16 Bt[128 * GBT_LDB];
  const int cntA = scal[3];
  const int rowBase = blockIdx.x * 128;
  const int t = threadIdx.x;
  const int w32 = t >> 5;
  const int lane = t & 31;
  const int mn = lane & 15, hi = lane >> 4;
  v8f acc[8];
#pragma unroll
  for (int i = 0; i < 8; ++i) acc[i] = v8f{0, 0, 0, 0, 0, 0, 0, 0};
  for (int c = 0; c < 6; ++c) {  // K = 384 in 6 chunks of 64
    const int seg = c >> 1;
    const int koff = (c & 1) * 64;
    if (c < 5) __builtin_prefetch(w + (c + 1) * 64 * D, 0, 0);
    // gather A chunk from the virtual concat [pf | cfa[inv] | amean[ainv[inv]]]
#pragma unroll
    for (int i = 0; i < 8; ++i) {
      int idx = i * 256 + t;
      int r = idx >> 4;
      int kk = (idx & 15) * 4;
      int gr = rowBase + r;
      float4 x = make_float4(0.f, 0.f, 0.f, 0.f);
      if (gr < n) {
        if (seg == 0) {
          x = *(const float4*)(pf + gr * D + koff + kk);
        } else {
          int ci = inv[gr];
          if (seg == 1) x = *(const float4*)(cfa + ci * D + koff + kk);
          else          x = *(const float4*)(amean + ainv[ci] * D + koff + kk);
        }
      }
      h4 hh = {(_Float16)x.x, (_Float16)x.y, (_Float16)x.z, (_Float16)x.w};
      *(h4*)(As + r * GA_LDA + kk) = hh;
    }
#pragma unroll
    for (int i = 0; i < 8; ++i) {
      int idx = i * 256 + t;
      int col = idx & 127;
      int kk = (idx >> 7) * 4;
      float a0 = w[(c * 64 + kk + 0) * D + col];
      float a1 = w[(c * 64 + kk + 1) * D + col];
      float a2 = w[(c * 64 + kk + 2) * D + col];
      float a3 = w[(c * 64 + kk + 3) * D + col];
      h4 hh = {(_Float16)a0, (_Float16)a1, (_Float16)a2, (_Float16)a3};
      *(h4*)(Bt + col * GBT_LDB + kk) = hh;
    }
    __syncthreads();
#pragma unroll
    for (int ki = 0; ki < 2; ++ki) {
      v16h a = load_a_frag(&As[(w32 * 16) * GA_LDA + ki * 32], GA_LDA, mn, hi);
#pragma unroll
      for (int nt = 0; nt < 8; ++nt) {
        v16h b = load_b_frag(&Bt[(nt * 16) * GBT_LDB + ki * 32], GBT_LDB, mn, hi);
        acc[nt] = __builtin_amdgcn_wmma_f32_16x16x32_f16(false, a, false, b, (short)0,
                                                         acc[nt], false, false);
      }
    }
    __syncthreads();
  }
#pragma unroll
  for (int nt = 0; nt < 8; ++nt) {
    int col = nt * 16 + mn;
    float bb = bias[col];
    float sc = g[col] * rsqrtf(v[col] + 1e-5f);
    float mm = m[col], bee = be[col];
#pragma unroll
    for (int r = 0; r < 8; ++r) {
      int row = rowBase + w32 * 16 + r + hi * 8;
      if (row < n) {
        float x = acc[nt][r] + bb;
        x = (x - mm) * sc + bee;
        x = (x >= 0.0f) ? x : 0.1f * x;
        bool sel = (cntA > 1) && (amask[row] != 0);
        out[row * D + col] = sel ? x : pf[row * D + col];
      }
    }
  }
}

// ---------------- host launcher ----------------
extern "C" void kernel_launch(void* const* d_in, const int* in_sizes, int n_in,
                              void* d_out, int out_size, void* d_ws, size_t ws_size,
                              hipStream_t stream) {
  const float* pf   = (const float*)d_in[0];
  const float* pc   = (const float*)d_in[1];
  const int*   bidx = (const int*)d_in[2];
  const int*   fg   = (const int*)d_in[3];
  const float* w_pf = (const float*)d_in[5];
  const float* b_pf = (const float*)d_in[6];
  const float* g_pf = (const float*)d_in[7];
  const float* be_pf= (const float*)d_in[8];
  const float* m_pf = (const float*)d_in[9];
  const float* v_pf = (const float*)d_in[10];
  const float* w_cp = (const float*)d_in[11];
  const float* b_cp = (const float*)d_in[12];
  const float* g_cp = (const float*)d_in[13];
  const float* be_cp= (const float*)d_in[14];
  const float* m_cp = (const float*)d_in[15];
  const float* v_cp = (const float*)d_in[16];
  const int n = in_sizes[2];

  char* ws = (char*)d_ws;
  size_t off = 0;
  auto take = [&](size_t bytes) -> char* {
    off = (off + 255) & ~(size_t)255;
    char* p = ws + off;
    off += bytes;
    return p;
  };
  unsigned* ranks1 = (unsigned*)take((size_t)RANGE1 * 4);
  unsigned* ranks2 = (unsigned*)take((size_t)RANGE2 * 4);
  unsigned* bsums  = (unsigned*)take(4096 * 4);
  int*   vkey  = (int*)take((size_t)n * 4);
  int*   inv   = (int*)take((size_t)n * 4);
  int*   uk    = (int*)take((size_t)n * 4);
  int*   akey  = (int*)take((size_t)n * 4);
  int*   ainv  = (int*)take((size_t)n * 4);
  int*   amask = (int*)take((size_t)n * 4);
  float* cnt   = (float*)take((size_t)n * 4);
  float* acnt  = (float*)take((size_t)n * 4);
  float* csum  = (float*)take((size_t)n * D * 4);   // becomes cmean in place
  float* cfa   = (float*)take((size_t)n * D * 4);
  float* asum  = (float*)take((size_t)n * D * 4);   // becomes amean in place
  int*   scal  = (int*)take(64);

  hipMemsetAsync(ranks1, 0, (size_t)RANGE1 * 4, stream);
  hipMemsetAsync(ranks2, 0, (size_t)RANGE2 * 4, stream);
  hipMemsetAsync(cnt, 0, (size_t)n * 4, stream);
  hipMemsetAsync(acnt, 0, (size_t)n * 4, stream);
  hipMemsetAsync(csum, 0, (size_t)n * D * 4, stream);
  hipMemsetAsync(asum, 0, (size_t)n * D * 4, stream);

  const int nbP = (n + 255) / 256;
  const int nbF = (n * D + 255) / 256;
  const int nbT = (n + 127) / 128;
  const int nb1 = (RANGE1 + SCAN_TILE - 1) / SCAN_TILE;
  const int nb2 = (RANGE2 + SCAN_TILE - 1) / SCAN_TILE;

  k_init_scal<<<1, 32, 0, stream>>>(scal);
  k_keys_mark<<<nbP, 256, 0, stream>>>(pc, bidx, vkey, ranks1, n);
  k_scan_local<<<nb1, SCAN_BLOCK, 0, stream>>>(ranks1, RANGE1, bsums);
  k_scan_bsums<<<1, SCAN_BLOCK, 0, stream>>>(bsums, nb1);
  k_scan_add<<<nb1, SCAN_BLOCK, 0, stream>>>(ranks1, RANGE1, bsums);
  k_total<<<1, 32, 0, stream>>>(ranks1, RANGE1, scal, 1);
  k_build_inv<<<nbP, 256, 0, stream>>>(vkey, ranks1, fg, inv, uk, cnt, scal, n);
  k_scatter_feat<<<nbF, 256, 0, stream>>>(pf, inv, fg, csum, n * D);
  k_cmean<<<nbF, 256, 0, stream>>>(csum, cnt, scal, n * D);
  k_gemm_cluster<<<nbT, 256, 0, stream>>>(csum, w_cp, b_cp, g_cp, be_cp, m_cp, v_cp,
                                          cfa, scal, n);
  k_akey_mark<<<nbP, 256, 0, stream>>>(uk, akey, ranks2, scal, n);
  k_scan_local<<<nb2, SCAN_BLOCK, 0, stream>>>(ranks2, RANGE2, bsums);
  k_scan_bsums<<<1, SCAN_BLOCK, 0, stream>>>(bsums, nb2);
  k_scan_add<<<nb2, SCAN_BLOCK, 0, stream>>>(ranks2, RANGE2, bsums);
  k_total<<<1, 32, 0, stream>>>(ranks2, RANGE2, scal, 2);
  k_build_ainv<<<nbP, 256, 0, stream>>>(akey, ranks2, ainv, scal, n);
  k_acnt<<<nbP, 256, 0, stream>>>(ainv, acnt, scal, n);
  k_asum<<<nbF, 256, 0, stream>>>(cfa, ainv, asum, scal, n * D);
  k_amean<<<nbF, 256, 0, stream>>>(asum, acnt, scal, n * D);
  k_amask<<<nbP, 256, 0, stream>>>(inv, ainv, cfa, asum, amask, scal, n);
  k_gemm_point<<<nbT, 256, 0, stream>>>(pf, cfa, asum, inv, ainv, w_pf, b_pf, g_pf,
                                        be_pf, m_pf, v_pf, amask, scal,
                                        (float*)d_out, n);
}
